// SoftDecisionTree_69904887709721
// MI455X (gfx1250) — matrix operand order repaired and checked by reading, not compiled, not run on hardware
//
#include <hip/hip_runtime.h>
#include <hip/hip_bf16.h>

typedef __attribute__((ext_vector_type(16))) _Float16 v16h;
typedef __attribute__((ext_vector_type(8)))  float    v8f;

#define INPUT_SIZE     128
#define NB_CLASSES     10
#define N_NODES        255
#define N_LEAVES       256
#define ROWS_PER_BLOCK 128
#define LDS_STRIDE     257   // 128*257*4B LDS, stride 257 -> conflict-free column access

// Workspace layout (bytes):
//   [0,      65536)  Wh   : f16 W, padded to 256 rows x 128
//   [65536,  66560)  r    : f32[256] leaf values (softmax(leaf_dist) @ class_reward)
//   [66560,  68608)  bsum : f32[512] per-block partial sums
#define WS_WH_OFF   0
#define WS_R_OFF    65536
#define WS_BSUM_OFF 66560

// ---------------------------------------------------------------------------
// Prep: convert W to f16 (zero-pad row 255), compute r[leaf].
// ---------------------------------------------------------------------------
__global__ void sdt_prep(const float* __restrict__ W,
                         const float* __restrict__ leaf_dist,
                         const float* __restrict__ class_reward,
                         _Float16* __restrict__ Wh,
                         float* __restrict__ r) {
    const int tid = threadIdx.x;           // 256 threads
    // f16 copy of W, row 255 zeroed (WMMA N-padding)
    {
        const float* src = W + (size_t)tid * INPUT_SIZE;
        _Float16*    dst = Wh + (size_t)tid * INPUT_SIZE;
        #pragma unroll 4
        for (int k = 0; k < INPUT_SIZE; ++k)
            dst[k] = (tid < N_NODES) ? (_Float16)src[k] : (_Float16)0.0f;
    }
    // r[leaf] = softmax(leaf_dist[leaf,:]) . class_reward
    {
        float v[NB_CLASSES];
        float m = -3.0e38f;
        #pragma unroll
        for (int c = 0; c < NB_CLASSES; ++c) {
            v[c] = leaf_dist[tid * NB_CLASSES + c];
            m = fmaxf(m, v[c]);
        }
        float s = 0.0f, acc = 0.0f;
        #pragma unroll
        for (int c = 0; c < NB_CLASSES; ++c) {
            float e = __expf(v[c] - m);
            s   += e;
            acc += e * class_reward[c];
        }
        r[tid] = acc / s;
    }
}

// ---------------------------------------------------------------------------
// Helpers for the WMMA pipeline
// ---------------------------------------------------------------------------
__device__ __forceinline__ void load_bfrag(const _Float16* base, v16h bf[4]) {
    #pragma unroll
    for (int k = 0; k < 4; ++k)
        bf[k] = *(const v16h*)(base + k * 32);
}

__device__ __forceinline__ v8f gemm4(const v16h a[4], const v16h bf[4]) {
    v8f c = {};
    #pragma unroll
    for (int k = 0; k < 4; ++k)
        c = __builtin_amdgcn_wmma_f32_16x16x32_f16(
                false, a[k], false, bf[k], (short)0, c, false, false);
    return c;
}

__device__ __forceinline__ void epilogue(v8f c, int node, int rb,
                                         const float* __restrict__ b,
                                         const float* __restrict__ beta,
                                         float* __restrict__ lds) {
    if (node < N_NODES) {
        const float bn = b[node];
        const float bt = beta[node];
        #pragma unroll
        for (int i = 0; i < 8; ++i) {
            const float z = bt * (c[i] + bn);
            const float e = __expf(-z);
            const float p = __builtin_amdgcn_rcpf(1.0f + e);   // v_rcp_f32, no div chain
            lds[(rb + i) * LDS_STRIDE + node] = p;
        }
    }
}

// ---------------------------------------------------------------------------
// Main: f16 WMMA GEMM (x @ W^T), sigmoid(beta*(.+b)) -> LDS, in-place tree
// fold per row, deterministic block reduction.
// 512 blocks x 256 threads (8 waves); wave w handles rows [w*16, w*16+16).
// ---------------------------------------------------------------------------
__global__ __launch_bounds__(256) void sdt_main(const float*    __restrict__ x,
                                                const float*    __restrict__ b,
                                                const float*    __restrict__ beta,
                                                const _Float16* __restrict__ Wh,
                                                const float*    __restrict__ rleaf,
                                                float*          __restrict__ blockSums) {
    extern __shared__ float lds[];          // [128][257] p-values, then fold values
    __shared__ float rl[N_LEAVES];          // staged leaf values
    __shared__ float partials[8];

    const int tid    = threadIdx.x;
    const int wave   = tid >> 5;
    const int lane   = tid & 31;
    const int hi16   = lane >> 4;           // 0: lanes 0-15, 1: lanes 16-31
    const int l15    = lane & 15;
    const int rowBase = blockIdx.x * ROWS_PER_BLOCK + wave * 16;

    rl[tid] = rleaf[tid];                   // completes before the post-GEMM barrier

    // ---- A fragments: 16x32 f16 per k-chunk, converted from f32 x ----
    // Layout (matches B): lanes 0-15 hold K = [0..15] of row M=lane;
    //             lanes 16-31 hold K = [16..31] of row M=lane-16.
    v16h a[4];
    {
        const float* xr = x + (size_t)(rowBase + l15) * INPUT_SIZE;
        const int s = hi16 ? 16 : 0;
        #pragma unroll
        for (int k = 0; k < 4; ++k) {
            const v8f s0 = *(const v8f*)(xr + k * 32 + s);
            const v8f s1 = *(const v8f*)(xr + k * 32 + s + 8);
            v16h av;
            #pragma unroll
            for (int i = 0; i < 8; ++i) {
                av[i]     = (_Float16)s0[i];
                av[i + 8] = (_Float16)s1[i];
            }
            a[k] = av;
        }
    }

    // ---- GEMM over 16 N-tiles, K = 4 chunks of 32; B double-buffered ----
    // B 32x16 layout: lane holds column n=lane%16, K = [0..15] (lanes 0-15)
    // or [16..31] (lanes 16-31): 16 contiguous f16 in W's row -> one 32B load.
    const int kb_off = hi16 ? 16 : 0;
    const int rb     = wave * 16 + (hi16 ? 8 : 0);
    const _Float16* wbase = Wh + (size_t)l15 * INPUT_SIZE + kb_off;

    v16h bfA[4], bfB[4];
    load_bfrag(wbase, bfA);                             // tile 0
    #pragma unroll
    for (int nt = 0; nt < 16; nt += 2) {
        load_bfrag(wbase + (size_t)(nt + 1) * 16 * INPUT_SIZE, bfB);   // prefetch nt+1
        v8f c = gemm4(a, bfA);
        epilogue(c, nt * 16 + l15, rb, b, beta, lds);
        if (nt + 2 < 16)
            load_bfrag(wbase + (size_t)(nt + 2) * 16 * INPUT_SIZE, bfA); // prefetch nt+2
        c = gemm4(a, bfB);
        epilogue(c, (nt + 1) * 16 + l15, rb, b, beta, lds);
    }
    __syncthreads();

    // ---- In-place bottom-up tree fold: one thread per row ----
    float val = 0.0f;
    if (tid < ROWS_PER_BLOCK) {
        float* rp = lds + (size_t)tid * LDS_STRIDE;
        for (int i = 254; i >= 127; --i) {         // depth-7 nodes: children are leaves
            const float pv = rp[i];
            val = pv * rl[2 * i - 254] + (1.0f - pv) * rl[2 * i - 253];
            rp[i] = val;
        }
        for (int i = 126; i >= 0; --i) {           // internal nodes
            const float pv = rp[i];
            val = pv * rp[2 * i + 1] + (1.0f - pv) * rp[2 * i + 2];
            rp[i] = val;
        }
        // val == rp[0] == per-row expected reward
    }

    // ---- Deterministic block reduction (wave32 shuffle tree + fixed order) ----
    #pragma unroll
    for (int off = 16; off > 0; off >>= 1)
        val += __shfl_xor(val, off, 32);
    if (lane == 0) partials[wave] = val;           // waves 4-7 contribute 0
    __syncthreads();
    if (tid == 0) {
        float s = 0.0f;
        #pragma unroll
        for (int w = 0; w < 8; ++w) s += partials[w];
        blockSums[blockIdx.x] = s;
    }
}

// ---------------------------------------------------------------------------
// Final fixed-order reduction of 512 block partials (deterministic).
// ---------------------------------------------------------------------------
__global__ void sdt_reduce(const float* __restrict__ bsum, float* __restrict__ out) {
    __shared__ float s[512];
    const int tid = threadIdx.x;
    s[tid] = bsum[tid];
    __syncthreads();
    for (int off = 256; off > 0; off >>= 1) {
        if (tid < off) s[tid] += s[tid + off];
        __syncthreads();
    }
    if (tid == 0) out[0] = s[0];
}

extern "C" void kernel_launch(void* const* d_in, const int* in_sizes, int n_in,
                              void* d_out, int out_size, void* d_ws, size_t ws_size,
                              hipStream_t stream) {
    const float* x    = (const float*)d_in[0];  // (65536,128)
    const float* W    = (const float*)d_in[1];  // (255,128)
    const float* b    = (const float*)d_in[2];  // (255,)
    const float* beta = (const float*)d_in[3];  // (255,)
    const float* leaf = (const float*)d_in[4];  // (256,10)
    const float* cr   = (const float*)d_in[5];  // (10,)

    char* ws = (char*)d_ws;
    _Float16* Wh   = (_Float16*)(ws + WS_WH_OFF);
    float*    r    = (float*)   (ws + WS_R_OFF);
    float*    bsum = (float*)   (ws + WS_BSUM_OFF);

    sdt_prep<<<1, 256, 0, stream>>>(W, leaf, cr, Wh, r);

    const size_t ldsBytes = (size_t)ROWS_PER_BLOCK * LDS_STRIDE * sizeof(float); // 128.5 KB
    const int    nBlocks  = 65536 / ROWS_PER_BLOCK;                              // 512
    sdt_main<<<nBlocks, 256, ldsBytes, stream>>>(x, b, beta, Wh, r, bsum);

    sdt_reduce<<<1, 512, 0, stream>>>(bsum, (float*)d_out);
}
